// OptimizedMambaBlock_85341000172236
// MI455X (gfx1250) — matrix operand order repaired and checked by reading, not compiled, not run on hardware
//
#include <hip/hip_runtime.h>
#include <stdint.h>

// ---------------- problem constants (from reference) ----------------
#define DIM_     256
#define D_INNER_ 512
#define N_STATE  16
#define DT_RANK_ 16
#define B_SZ     2
#define HW_      2304                 // 48*48
#define M_TOT    (B_SZ * HW_)         // 4608
#define SCAN_CHUNK 64

// ---------------- CDNA5 WMMA types ----------------
typedef __attribute__((ext_vector_type(16))) __bf16 v16bf;
typedef __attribute__((ext_vector_type(8)))  __bf16 v8bf;
typedef __attribute__((ext_vector_type(8)))  float  v8f;

__device__ __forceinline__ v8f wmma_bf16(v16bf a, v16bf b, v8f c) {
  // D = A(16x32 bf16) * B(32x16 bf16) + C(16x16 f32)
  return __builtin_amdgcn_wmma_f32_16x16x32_bf16(false, a, false, b, (short)0, c,
                                                 false, false);
}

// A fragment: row-major activation [M][K] bf16.
// lane: row = m0 + (lane&15); elems 0..7 -> K = k0 + (lane>>4)*16 + 0..7
//                             elems 8..15 -> same + 8
__device__ __forceinline__ v16bf load_a_frag(const __bf16* __restrict__ A, int lda,
                                             int m0, int k0, int lane) {
  const int row = m0 + (lane & 15);
  const int kb  = k0 + ((lane >> 4) << 4);
  v8bf lo = *reinterpret_cast<const v8bf*>(A + (size_t)row * lda + kb);
  v8bf hi = *reinterpret_cast<const v8bf*>(A + (size_t)row * lda + kb + 8);
  v16bf a;
#pragma unroll
  for (int i = 0; i < 8; ++i) { a[i] = lo[i]; a[i + 8] = hi[i]; }
  return a;
}

// B fragment: weight stored row-major [N][K] bf16 (y = x * W^T).
// lane: col = n0 + (lane&15); elems 0..15 -> K = k0 + (lane>>4)*16 + 0..15
__device__ __forceinline__ v16bf load_b_frag(const __bf16* __restrict__ W, int ldb,
                                             int n0, int k0, int lane) {
  const int col = n0 + (lane & 15);
  const int kb  = k0 + ((lane >> 4) << 4);
  return *reinterpret_cast<const v16bf*>(W + (size_t)col * ldb + kb);
}

__device__ __forceinline__ float softplusf(float v) {
  return (v > 20.0f) ? v : log1pf(__expf(v));
}
__device__ __forceinline__ float siluf(float v) {
  return v / (1.0f + __expf(-v));
}

// ---------------- kernel 0: weights fp32 -> bf16 (dt_proj zero-padded K 16->32) --
__global__ void k_convert_weights(const float* __restrict__ inW,
                                  const float* __restrict__ outW,
                                  const float* __restrict__ xW,
                                  const float* __restrict__ dtW,
                                  __bf16* __restrict__ inWb, __bf16* __restrict__ outWb,
                                  __bf16* __restrict__ xWb, __bf16* __restrict__ dtWb) {
  const int i = blockIdx.x * blockDim.x + threadIdx.x;
  if (i < 1024 * 256) inWb[i]  = (__bf16)inW[i];
  if (i < 256 * 512)  outWb[i] = (__bf16)outW[i];
  if (i < 48 * 512)   xWb[i]   = (__bf16)xW[i];
  if (i < 512 * 32) {
    const int r = i >> 5, c = i & 31;
    dtWb[i] = (c < DT_RANK_) ? (__bf16)dtW[r * DT_RANK_ + c] : (__bf16)0.0f;
  }
}

// ---------------- kernel 1: channel LayerNorm -> bf16 seq [M_TOT][256] ----------
__global__ void k_layernorm(const float* __restrict__ x, const float* __restrict__ w,
                            const float* __restrict__ bias, __bf16* __restrict__ seq) {
  const int row = blockIdx.x;          // b*HW + l
  const int c   = threadIdx.x;         // 0..255
  const int b   = row / HW_;
  const int l   = row - b * HW_;
  const float v = x[((size_t)b * DIM_ + c) * HW_ + l];
  __shared__ float s1[DIM_], s2[DIM_];
  s1[c] = v; s2[c] = v * v;
  __syncthreads();
  for (int s = DIM_ / 2; s > 0; s >>= 1) {
    if (c < s) { s1[c] += s1[c + s]; s2[c] += s2[c + s]; }
    __syncthreads();
  }
  const float mu  = s1[0] * (1.0f / DIM_);
  const float var = s2[0] * (1.0f / DIM_) - mu * mu;
  const float rs  = rsqrtf(var + 1e-5f);
  seq[(size_t)row * DIM_ + c] = (__bf16)((v - mu) * rs * w[c] + bias[c]);
}

// ---------------- kernel 2: in_proj GEMM  [4608,256] x [256,1024] -> xz f32 -----
// Register double-buffered: next k-step's fragments are in flight while the
// current step's 4 WMMAs issue, so the s_wait before WMMA finds loads drained.
__global__ void k_gemm_in_proj(const __bf16* __restrict__ A, const __bf16* __restrict__ W,
                               float* __restrict__ C) {
  const int lane = threadIdx.x & 31;
  const int wave = threadIdx.x >> 5;             // 8 waves stacked in M
  const int m0 = blockIdx.x * 128 + wave * 16;
  const int n0 = blockIdx.y * 64;
  v8f acc[4] = {};
  v16bf aC = load_a_frag(A, DIM_, m0, 0, lane);
  v16bf bC[4];
#pragma unroll
  for (int j = 0; j < 4; ++j) bC[j] = load_b_frag(W, DIM_, n0 + 16 * j, 0, lane);
#pragma unroll
  for (int k0 = 0; k0 < DIM_ - 32; k0 += 32) {
    const int kn = k0 + 32;
    v16bf aN = load_a_frag(A, DIM_, m0, kn, lane);
    v16bf bN[4];
#pragma unroll
    for (int j = 0; j < 4; ++j) bN[j] = load_b_frag(W, DIM_, n0 + 16 * j, kn, lane);
#pragma unroll
    for (int j = 0; j < 4; ++j) acc[j] = wmma_bf16(aC, bC[j], acc[j]);
    aC = aN;
#pragma unroll
    for (int j = 0; j < 4; ++j) bC[j] = bN[j];
  }
#pragma unroll
  for (int j = 0; j < 4; ++j) acc[j] = wmma_bf16(aC, bC[j], acc[j]);

  const int rbase = m0 + ((lane >> 4) << 3);
  const int cb    = n0 + (lane & 15);
#pragma unroll
  for (int j = 0; j < 4; ++j)
#pragma unroll
    for (int r = 0; r < 8; ++r)
      C[(size_t)(rbase + r) * 1024 + cb + 16 * j] = acc[j][r];
}

// ---------------- kernel 3: depthwise causal conv (k=4) + SiLU ------------------
__global__ void k_conv_silu(const float* __restrict__ xz, const float* __restrict__ cw,
                            const float* __restrict__ cb, float* __restrict__ xu,
                            __bf16* __restrict__ xub) {
  const int i = blockIdx.x * blockDim.x + threadIdx.x;
  if (i >= M_TOT * D_INNER_) return;
  const int row = i >> 9;
  const int d   = i & 511;
  const int l   = row % HW_;
  float acc = cb[d];
#pragma unroll
  for (int j = 0; j < 4; ++j)                    // tap x[l-j] * w[3-j]
    if (l - j >= 0) acc = fmaf(xz[(size_t)(row - j) * 1024 + d], cw[d * 4 + (3 - j)], acc);
  const float s = siluf(acc);
  xu[i]  = s;
  xub[i] = (__bf16)s;
}

// ---------------- kernel 4: x_proj GEMM [4608,512] x [512,48] -> dt/B/C ---------
__global__ void k_gemm_x_proj(const __bf16* __restrict__ A, const __bf16* __restrict__ W,
                              __bf16* __restrict__ dtb, float* __restrict__ Bm,
                              float* __restrict__ Cm) {
  const int lane = threadIdx.x & 31;
  const int wave = threadIdx.x >> 5;
  const int m0 = blockIdx.x * 128 + wave * 16;
  v8f acc[3] = {};
  v16bf aC = load_a_frag(A, D_INNER_, m0, 0, lane);
  v16bf bC[3];
#pragma unroll
  for (int j = 0; j < 3; ++j) bC[j] = load_b_frag(W, D_INNER_, 16 * j, 0, lane);
#pragma unroll 5
  for (int k0 = 0; k0 < D_INNER_ - 32; k0 += 32) {
    const int kn = k0 + 32;
    v16bf aN = load_a_frag(A, D_INNER_, m0, kn, lane);
    v16bf bN[3];
#pragma unroll
    for (int j = 0; j < 3; ++j) bN[j] = load_b_frag(W, D_INNER_, 16 * j, kn, lane);
#pragma unroll
    for (int j = 0; j < 3; ++j) acc[j] = wmma_bf16(aC, bC[j], acc[j]);
    aC = aN;
#pragma unroll
    for (int j = 0; j < 3; ++j) bC[j] = bN[j];
  }
#pragma unroll
  for (int j = 0; j < 3; ++j) acc[j] = wmma_bf16(aC, bC[j], acc[j]);

  const int rbase = m0 + ((lane >> 4) << 3);
  const int col   = lane & 15;
#pragma unroll
  for (int r = 0; r < 8; ++r) {
    const int row = rbase + r;
    dtb[row * 32 + col]      = (__bf16)acc[0][r];   // dt, K-padded with zeros
    dtb[row * 32 + col + 16] = (__bf16)0.0f;
    Bm[row * 16 + col] = acc[1][r];
    Cm[row * 16 + col] = acc[2][r];
  }
}

// ---------------- kernel 5: dt_proj GEMM [4608,32] x [32,512] + softplus --------
__global__ void k_gemm_dt_proj(const __bf16* __restrict__ A, const __bf16* __restrict__ W,
                               const float* __restrict__ bias, float* __restrict__ delta) {
  const int lane = threadIdx.x & 31;
  const int wave = threadIdx.x >> 5;
  const int m0 = blockIdx.x * 128 + wave * 16;
  const int n0 = blockIdx.y * 64;
  v16bf a = load_a_frag(A, 32, m0, 0, lane);
  v8f acc[4] = {};
#pragma unroll
  for (int j = 0; j < 4; ++j)
    acc[j] = wmma_bf16(a, load_b_frag(W, 32, n0 + 16 * j, 0, lane), acc[j]);
  const int rbase = m0 + ((lane >> 4) << 3);
#pragma unroll
  for (int j = 0; j < 4; ++j) {
    const int c = n0 + 16 * j + (lane & 15);
    const float bb = bias[c];
#pragma unroll
    for (int r = 0; r < 8; ++r)
      delta[(size_t)(rbase + r) * D_INNER_ + c] = softplusf(acc[j][r] + bb);
  }
}

// ---------------- kernel 6: selective scan (sequential over L) + gate -----------
// One thread per (b, d) channel.  The per-chunk B/C state vectors (4 KB each) are
// staged global->LDS with GLOBAL_LOAD_ASYNC_TO_LDS_B128 (ASYNCcnt path, no VGPR
// round-trip), drained with s_wait_asynccnt before the workgroup barrier.
__global__ void k_scan(const float* __restrict__ delta, const float* __restrict__ xu,
                       const float* __restrict__ xz, const float* __restrict__ Bm,
                       const float* __restrict__ Cm, const float* __restrict__ A_log,
                       const float* __restrict__ Dp, __bf16* __restrict__ yB) {
  const int tid = threadIdx.x;
  const int b   = blockIdx.x >> 1;
  const int d   = ((blockIdx.x & 1) << 8) + tid;
  __shared__ float sB[SCAN_CHUNK * 16];
  __shared__ float sC[SCAN_CHUNK * 16];
  float Areg[16], h[16];
#pragma unroll
  for (int n = 0; n < 16; ++n) { Areg[n] = -__expf(A_log[d * 16 + n]); h[n] = 0.0f; }
  const float Dd = Dp[d];
  const int rowBase = b * HW_;
  const uint32_t ldsB = (uint32_t)(uintptr_t)(&sB[tid * 4]);
  const uint32_t ldsC = (uint32_t)(uintptr_t)(&sC[tid * 4]);
  for (int c0 = 0; c0 < HW_; c0 += SCAN_CHUNK) {
    __syncthreads();   // previous chunk fully consumed before LDS is overwritten
    {
      const float* gB = &Bm[(size_t)(rowBase + c0) * 16 + tid * 4];
      const float* gC = &Cm[(size_t)(rowBase + c0) * 16 + tid * 4];
      asm volatile("global_load_async_to_lds_b128 %0, %1, off"
                   :: "v"(ldsB), "v"(gB) : "memory");
      asm volatile("global_load_async_to_lds_b128 %0, %1, off"
                   :: "v"(ldsC), "v"(gC) : "memory");
      asm volatile("s_wait_asynccnt 0" ::: "memory");
    }
    if (c0 + SCAN_CHUNK < HW_) {   // hint next chunk toward L2
      __builtin_prefetch(&Bm[(size_t)(rowBase + c0 + SCAN_CHUNK) * 16], 0, 0);
      __builtin_prefetch(&Cm[(size_t)(rowBase + c0 + SCAN_CHUNK) * 16], 0, 0);
    }
    __syncthreads();
    for (int ll = 0; ll < SCAN_CHUNK; ++ll) {
      const int row = rowBase + c0 + ll;
      const float dl = delta[(size_t)row * D_INNER_ + d];
      const float xv = xu[(size_t)row * D_INNER_ + d];
      const float dx = dl * xv;
      float y = 0.0f;
#pragma unroll
      for (int n = 0; n < 16; ++n) {
        h[n] = __expf(dl * Areg[n]) * h[n] + dx * sB[ll * 16 + n];
        y = fmaf(h[n], sC[ll * 16 + n], y);
      }
      y = fmaf(xv, Dd, y);
      const float z = xz[(size_t)row * 1024 + D_INNER_ + d];
      yB[(size_t)row * D_INNER_ + d] = (__bf16)(y * siluf(z));
    }
  }
}

// ---------------- kernel 7: out_proj GEMM + transpose + residual ----------------
__global__ void k_gemm_out_proj(const __bf16* __restrict__ A, const __bf16* __restrict__ W,
                                const float* __restrict__ xres,
                                const float* __restrict__ gamma, float* __restrict__ out) {
  const int lane = threadIdx.x & 31;
  const int wave = threadIdx.x >> 5;
  const int m0 = blockIdx.x * 128 + wave * 16;
  const int n0 = blockIdx.y * 64;
  v8f acc[4] = {};
  v16bf aC = load_a_frag(A, D_INNER_, m0, 0, lane);
  v16bf bC[4];
#pragma unroll
  for (int j = 0; j < 4; ++j) bC[j] = load_b_frag(W, D_INNER_, n0 + 16 * j, 0, lane);
#pragma unroll 5
  for (int k0 = 0; k0 < D_INNER_ - 32; k0 += 32) {
    const int kn = k0 + 32;
    v16bf aN = load_a_frag(A, D_INNER_, m0, kn, lane);
    v16bf bN[4];
#pragma unroll
    for (int j = 0; j < 4; ++j) bN[j] = load_b_frag(W, D_INNER_, n0 + 16 * j, kn, lane);
#pragma unroll
    for (int j = 0; j < 4; ++j) acc[j] = wmma_bf16(aC, bC[j], acc[j]);
    aC = aN;
#pragma unroll
    for (int j = 0; j < 4; ++j) bC[j] = bN[j];
  }
#pragma unroll
  for (int j = 0; j < 4; ++j) acc[j] = wmma_bf16(aC, bC[j], acc[j]);

  const float g   = gamma[0];
  const int rbase = m0 + ((lane >> 4) << 3);
  const int bb    = rbase / HW_;               // whole tile shares b (HW_%16==0)
  const int lbase = rbase - bb * HW_;
#pragma unroll
  for (int j = 0; j < 4; ++j) {
    const int c = n0 + 16 * j + (lane & 15);
    const size_t base = ((size_t)bb * DIM_ + c) * HW_ + lbase;
#pragma unroll
    for (int r = 0; r < 8; ++r)
      out[base + r] = xres[base + r] + g * acc[j][r];
  }
}

// ---------------- launcher ----------------
extern "C" void kernel_launch(void* const* d_in, const int* in_sizes, int n_in,
                              void* d_out, int out_size, void* d_ws, size_t ws_size,
                              hipStream_t stream) {
  const float* x         = (const float*)d_in[0];
  const float* ln_w      = (const float*)d_in[1];
  const float* ln_b      = (const float*)d_in[2];
  const float* in_proj_w = (const float*)d_in[3];
  const float* conv_w    = (const float*)d_in[4];
  const float* conv_b    = (const float*)d_in[5];
  const float* x_proj_w  = (const float*)d_in[6];
  const float* dt_proj_w = (const float*)d_in[7];
  const float* dt_proj_b = (const float*)d_in[8];
  const float* A_log     = (const float*)d_in[9];
  const float* Dvec      = (const float*)d_in[10];
  const float* out_proj_w= (const float*)d_in[11];
  const float* gamma     = (const float*)d_in[12];
  float* out = (float*)d_out;

  char* ws = (char*)d_ws;
  // byte offsets (all 256-aligned by construction)
  __bf16* seqB   = (__bf16*)(ws + 0);          //  4608*256  bf16
  __bf16* wInB   = (__bf16*)(ws + 2359296);    //  1024*256  bf16
  __bf16* wOutB  = (__bf16*)(ws + 2883584);    //  256*512   bf16
  __bf16* wXB    = (__bf16*)(ws + 3145728);    //  48*512    bf16
  __bf16* wDtB   = (__bf16*)(ws + 3194880);    //  512*32    bf16 (padded)
  float*  xz     = (float*) (ws + 3227648);    //  4608*1024 f32
  float*  xu     = (float*) (ws + 22102016);   //  4608*512  f32
  __bf16* xuB    = (__bf16*)(ws + 31539200);   //  4608*512  bf16
  __bf16* dtB    = (__bf16*)(ws + 36257792);   //  4608*32   bf16 (padded)
  float*  BmBuf  = (float*) (ws + 36552704);   //  4608*16   f32
  float*  CmBuf  = (float*) (ws + 36847616);   //  4608*16   f32
  float*  deltaB = (float*) (ws + 37142528);   //  4608*512  f32
  __bf16* yBuf   = (__bf16*)(ws + 46579712);   //  4608*512  bf16
  // total ws usage: 51,298,304 bytes

  k_convert_weights<<<1024, 256, 0, stream>>>(in_proj_w, out_proj_w, x_proj_w,
                                              dt_proj_w, wInB, wOutB, wXB, wDtB);
  k_layernorm<<<M_TOT, 256, 0, stream>>>(x, ln_w, ln_b, seqB);
  k_gemm_in_proj<<<dim3(M_TOT / 128, 1024 / 64), 256, 0, stream>>>(seqB, wInB, xz);
  k_conv_silu<<<(M_TOT * D_INNER_) / 256, 256, 0, stream>>>(xz, conv_w, conv_b, xu, xuB);
  k_gemm_x_proj<<<M_TOT / 128, 256, 0, stream>>>(xuB, wXB, dtB, BmBuf, CmBuf);
  k_gemm_dt_proj<<<dim3(M_TOT / 128, D_INNER_ / 64), 256, 0, stream>>>(dtB, wDtB,
                                                                      dt_proj_b, deltaB);
  k_scan<<<B_SZ * 2, 256, 0, stream>>>(deltaB, xu, xz, BmBuf, CmBuf, A_log, Dvec, yBuf);
  k_gemm_out_proj<<<dim3(M_TOT / 128, DIM_ / 64), 256, 0, stream>>>(yBuf, wOutB, x,
                                                                   gamma, out);
}
